// PointConvDis_76175539962312
// MI455X (gfx1250) — matrix operand order repaired and checked by reading, not compiled, not run on hardware
//
#include <hip/hip_runtime.h>

typedef __attribute__((ext_vector_type(16))) _Float16 v16h;
typedef __attribute__((ext_vector_type(8)))  _Float16 v8h;
typedef __attribute__((ext_vector_type(8)))  float    v8f;

#define Bsz   8
#define Cch   64
#define Npts  16384
#define Kn    8
#define INCH  520      // 8 * 65
#define KT1   17       // layer-1 K tiles (544 padded channels / 32)
#define H1    128
#define H3    256

// weight-fragment layout in d_ws (units: _Float16 halves)
#define W1F_OFF    0
#define W1F_HALVES (8 * KT1 * 512)          // 69632
#define W2F_OFF    (W1F_OFF + W1F_HALVES)
#define W2F_HALVES (8 * 4 * 512)            // 16384
#define W3F_OFF    (W2F_OFF + W2F_HALVES)
#define W3F_HALVES (16 * 4 * 512)           // 32768
#define TOTAL_HALVES (W3F_OFF + W3F_HALVES) // 118784

// A-fragment K index for 16-bit 16x32 A matrix (matches B-fragment order):
// lanes 0-15: halves 0-15 -> K=0-15; lanes 16-31: K=16-31
__device__ __forceinline__ int kmapA(int lane, int h) {
    return h + ((lane & 16) ? 16 : 0);
}

// B-fragment slot for channel ch, point p (16-bit 32x16 B matrix):
// lanes 0-15 hold K=0-15 (half h -> K=h), lanes 16-31 hold K=16-31; N = lane%16
__device__ __forceinline__ int bslot(int ch, int p) {
    int r = ch & 31;
    int lane = (r & 16) ? (p + 16) : p;
    return (ch >> 5) * 512 + lane * 16 + (r & 15);
}

__global__ __launch_bounds__(256)
void pcv_prep_weights(const float* __restrict__ W1,
                      const float* __restrict__ W2,
                      const float* __restrict__ W3,
                      _Float16* __restrict__ wf) {
    int i = blockIdx.x * blockDim.x + threadIdx.x;
    if (i >= TOTAL_HALVES) return;
    float v;
    if (i < W1F_HALVES) {
        int mtkt = i >> 9, within = i & 511;
        int lane = within >> 4, h = within & 15;
        int mt = mtkt / KT1, kt = mtkt - mt * KT1;
        int m  = mt * 16 + (lane & 15);
        int ch = kt * 32 + kmapA(lane, h);
        v = (ch < INCH) ? W1[m * INCH + ch] : 0.0f;
    } else if (i < W2F_OFF + W2F_HALVES) {
        int j = i - W2F_OFF;
        int mtkt = j >> 9, within = j & 511;
        int lane = within >> 4, h = within & 15;
        int mt = mtkt >> 2, kt = mtkt & 3;
        int m  = mt * 16 + (lane & 15);
        int ch = kt * 32 + kmapA(lane, h);
        v = W2[m * H1 + ch];
    } else {
        int j = i - W3F_OFF;
        int mtkt = j >> 9, within = j & 511;
        int lane = within >> 4, h = within & 15;
        int mt = mtkt >> 2, kt = mtkt & 3;
        int m  = mt * 16 + (lane & 15);
        int ch = kt * 32 + kmapA(lane, h);
        v = W3[m * H1 + ch];
    }
    wf[i] = (_Float16)v;
}

__global__ __launch_bounds__(256)
void pcv_fused_mlp(const float* __restrict__ feature,
                   const int*   __restrict__ idx,
                   const float* __restrict__ dis,
                   const _Float16* __restrict__ wf,
                   const float* __restrict__ b1,
                   const float* __restrict__ b2,
                   const float* __restrict__ b3,
                   float* __restrict__ out) {
    __shared__ _Float16 sX0[KT1 * 512];  // 17.0 KB: 544ch x 16pts, B-frag order
    __shared__ _Float16 sX1[4 * 512];    //  4 KB: 128ch x 16pts
    __shared__ _Float16 sX2[4 * 512];    //  4 KB: 128ch x 16pts

    const int tid = threadIdx.x;
    const int tilesPerB = Npts / 16;
    const int t  = blockIdx.x;
    const int b  = t / tilesPerB;
    const int n0 = (t - b * tilesPerB) * 16;

    // ---- zero the pad channels 520..543 (24 ch x 16 pts = 384 slots) ----
    for (int s = tid; s < 384; s += 256)
        sX0[bslot(520 + (s >> 4), s & 15)] = (_Float16)0.0f;

    // ---- cooperative gather: thread = (point p, channel-chunk) ----
    const int p      = tid & 15;
    const int cchunk = tid >> 4;           // 0..15
    const int n      = n0 + p;
    const int* ip = idx + ((size_t)b * Npts + n) * Kn;
    int nbr[Kn];
#pragma unroll
    for (int k = 0; k < Kn; ++k) nbr[k] = ip[k];

    if (cchunk == 0) {                     // channel c'==0 is the distance
        const float* dp = dis + ((size_t)b * Npts + n) * Kn;
#pragma unroll
        for (int k = 0; k < Kn; ++k)
            sX0[bslot(k * 65, p)] = (_Float16)dp[k];
    }
    const float* fb = feature + (size_t)b * Cch * Npts;
#pragma unroll
    for (int j = 0; j < 4; ++j) {
        int c = cchunk + 16 * j;           // feature channel 0..63
        const float* fc = fb + (size_t)c * Npts;
#pragma unroll
        for (int k = 0; k < Kn; ++k) {
            float v = fc[nbr[k]];          // random gather, L2-resident slab
            sX0[bslot(k * 65 + 1 + c, p)] = (_Float16)v;
        }
    }
    __syncthreads();

    const int wave   = tid >> 5;           // wave32: 8 waves / block
    const int lane   = tid & 31;
    const int laneHi = (lane & 16) ? 8 : 0;
    const int np     = lane & 15;
    const int m0     = wave * 16 + laneHi; // base out-channel this lane holds
    const int lane2  = ((m0 & 31) < 16) ? np : np + 16;
    const int packSlotBase = lane2 * 16 + (m0 & 15);   // 8 contiguous halves

    // ---- layer 1: 520(->544) -> 128, ReLU ----
    v8f acc = {};
    const _Float16* w1f = wf + W1F_OFF;
#pragma unroll
    for (int kt = 0; kt < KT1; ++kt) {
        v16h a  = *(const v16h*)(w1f + ((size_t)(wave * KT1 + kt) * 512 + lane * 16));
        v16h bm = *(const v16h*)(sX0 + kt * 512 + lane * 16);
        acc = __builtin_amdgcn_wmma_f32_16x16x32_f16(false, a, false, bm,
                                                     (short)0, acc, false, false);
    }
    {
        v8h pk;
#pragma unroll
        for (int r = 0; r < 8; ++r) {
            float v = acc[r] + b1[m0 + r];
            pk[r] = (_Float16)(v > 0.0f ? v : 0.0f);
        }
        *(v8h*)(sX1 + (m0 >> 5) * 512 + packSlotBase) = pk;   // 1x ds_store_b128
    }
    __syncthreads();

    // ---- layer 2: 128 -> 128, ReLU ----
    v8f acc2 = {};
    const _Float16* w2f = wf + W2F_OFF;
#pragma unroll
    for (int kt = 0; kt < 4; ++kt) {
        v16h a  = *(const v16h*)(w2f + ((wave * 4 + kt) * 512 + lane * 16));
        v16h bm = *(const v16h*)(sX1 + kt * 512 + lane * 16);
        acc2 = __builtin_amdgcn_wmma_f32_16x16x32_f16(false, a, false, bm,
                                                      (short)0, acc2, false, false);
    }
    {
        v8h pk;
#pragma unroll
        for (int r = 0; r < 8; ++r) {
            float v = acc2[r] + b2[m0 + r];
            pk[r] = (_Float16)(v > 0.0f ? v : 0.0f);
        }
        *(v8h*)(sX2 + (m0 >> 5) * 512 + packSlotBase) = pk;
    }
    __syncthreads();

    // ---- layer 3: 128 -> 256, no activation, write f32 output ----
    const _Float16* w3f = wf + W3F_OFF;
    float* ob = out + (size_t)b * H3 * Npts + n0 + np;
#pragma unroll
    for (int half = 0; half < 2; ++half) {
        int mt = wave + half * 8;
        v8f acc3 = {};
#pragma unroll
        for (int kt = 0; kt < 4; ++kt) {
            v16h a  = *(const v16h*)(w3f + ((mt * 4 + kt) * 512 + lane * 16));
            v16h bm = *(const v16h*)(sX2 + kt * 512 + lane * 16);
            acc3 = __builtin_amdgcn_wmma_f32_16x16x32_f16(false, a, false, bm,
                                                          (short)0, acc3, false, false);
        }
#pragma unroll
        for (int r = 0; r < 8; ++r) {
            int m = mt * 16 + r + laneHi;
            ob[(size_t)m * Npts] = acc3[r] + b3[m];   // 16 pts coalesced per row
        }
    }
}

extern "C" void kernel_launch(void* const* d_in, const int* in_sizes, int n_in,
                              void* d_out, int out_size, void* d_ws, size_t ws_size,
                              hipStream_t stream) {
    const float* feature = (const float*)d_in[0];
    const int*   idx     = (const int*)  d_in[1];
    const float* dis     = (const float*)d_in[2];
    const float* W1      = (const float*)d_in[3];
    const float* b1      = (const float*)d_in[4];
    const float* W2      = (const float*)d_in[5];
    const float* b2      = (const float*)d_in[6];
    const float* W3      = (const float*)d_in[7];
    const float* b3      = (const float*)d_in[8];
    float*       out     = (float*)d_out;
    _Float16*    wf      = (_Float16*)d_ws;

    int prepBlocks = (TOTAL_HALVES + 255) / 256;
    pcv_prep_weights<<<prepBlocks, 256, 0, stream>>>(W1, W2, W3, wf);

    int tiles = Bsz * (Npts / 16);   // 8192 blocks, one 16-point tile each
    pcv_fused_mlp<<<tiles, 256, 0, stream>>>(feature, idx, dis, wf,
                                             b1, b2, b3, out);
}
